// GIN_ModelBen2_45792941310040
// MI455X (gfx1250) — compile-verified, matrix operand order checked by reading, not run on hardware
//
#include <hip/hip_runtime.h>
#include <stdint.h>

// ---------------------------------------------------------------------------
// GIN forward on gfx1250: bf16 WMMA GEMMs, LDS-staged A panels with fused
// f32->bf16 conversion, fragment-shuffled weights, f32 atomic scatter.
// ---------------------------------------------------------------------------

#define N_NODES 100000
#define N_EDGES 600000
#define IN_DIM  128
#define HID_DIM 512
#define OUT_DIM 40
#define OUT_PAD 48   // pad logits cols to 3 x 16 tiles

typedef __attribute__((ext_vector_type(16))) __bf16 v16bf;
typedef __attribute__((ext_vector_type(8)))  float  v8f;

union Frag16 {
    v16bf v;
    unsigned short u[16];
    uint4 q[2];
};

__device__ __forceinline__ unsigned short f2bf(float f) {
    return __builtin_bit_cast(unsigned short, static_cast<__bf16>(f));
}
__device__ __forceinline__ float bf2f(unsigned short h) {
    return __uint_as_float(((unsigned int)h) << 16);
}

// ---------------------------------------------------------------------------
// Utility kernels
// ---------------------------------------------------------------------------

__global__ void k_copy_f4(const float* __restrict__ in, float* __restrict__ out,
                          long long n4) {
    long long t = (long long)blockIdx.x * blockDim.x + threadIdx.x;
    if (t < n4) ((float4*)out)[t] = ((const float4*)in)[t];
}

// Shuffle row-major f32 W[K][NcolsSrc] into the WMMA B-fragment bf16 layout,
// padded to Ncols:  Ws[(((tn*(K/32)+ks)*32+lane)*16 + i]
//   = bf16( W[ ks*32 + kb + (i<8 ? i : i+8) ][ tn*16 + (lane&15) ] )
__global__ void k_shuffle_w(const float* __restrict__ W,
                            unsigned short* __restrict__ Ws,
                            int K, int Ncols, int NcolsSrc) {
    int t = blockIdx.x * blockDim.x + threadIdx.x;
    int nKs = K / 32;
    int total = (Ncols / 16) * nKs * 32;
    if (t >= total) return;
    int lane = t & 31;
    int ks   = (t >> 5) % nKs;
    int n  = ((t >> 5) / nKs) * 16 + (lane & 15);
    int kb = (lane < 16) ? 0 : 8;
    Frag16 f;
#pragma unroll
    for (int i = 0; i < 8; ++i) {
        int k1 = ks * 32 + kb + i;
        int k2 = k1 + 16;
        f.u[i]     = (n < NcolsSrc) ? f2bf(W[(size_t)k1 * NcolsSrc + n]) : (unsigned short)0;
        f.u[8 + i] = (n < NcolsSrc) ? f2bf(W[(size_t)k2 * NcolsSrc + n]) : (unsigned short)0;
    }
    uint4* dst = (uint4*)(Ws + (size_t)t * 16);
    dst[0] = f.q[0];
    dst[1] = f.q[1];
}

// ---------------------------------------------------------------------------
// Scatter-add aggregations (segment_sum via f32 atomics)
// ---------------------------------------------------------------------------

// agg1[dst] += x[src], 128 feats: 32 lanes/edge, 4 floats/lane
__global__ void k_scatter1(const float* __restrict__ x,
                           const long long* __restrict__ ei,
                           float* __restrict__ agg) {
    long long t = (long long)blockIdx.x * blockDim.x + threadIdx.x;
    long long e = t >> 5;
    if (e >= N_EDGES) return;
    int f = (int)(t & 31) * 4;
    long long s = ei[e];
    long long d = ei[N_EDGES + e];
    float4 v = *(const float4*)(x + s * IN_DIM + f);
    float* p = agg + d * IN_DIM + f;
    atomicAdd(p + 0, v.x); atomicAdd(p + 1, v.y);
    atomicAdd(p + 2, v.z); atomicAdd(p + 3, v.w);
}

// agg2[dst] += h[src], 512 feats (h stored bf16): 128 threads/edge, 4 vals each
__global__ void k_scatter2(const unsigned short* __restrict__ h,
                           const long long* __restrict__ ei,
                           float* __restrict__ agg) {
    long long t = (long long)blockIdx.x * blockDim.x + threadIdx.x;
    long long e = t >> 7;
    if (e >= N_EDGES) return;
    int f = (int)(t & 127) * 4;
    long long s = ei[e];
    long long d = ei[N_EDGES + e];
    ushort4 v = *(const ushort4*)(h + s * HID_DIM + f);
    float* p = agg + d * HID_DIM + f;
    atomicAdd(p + 0, bf2f(v.x)); atomicAdd(p + 1, bf2f(v.y));
    atomicAdd(p + 2, bf2f(v.z)); atomicAdd(p + 3, bf2f(v.w));
}

// ---------------------------------------------------------------------------
// GEMM 1: h = relu(agg1 @ W1 + b1).
// Block (8 waves) covers 2 row-tiles x all 32 N-tiles. The 32x128 f32 A panel
// is cooperatively loaded, converted to bf16 once, and staged in LDS (8 KB);
// waves build A fragments with two ds_load_b128. agg1 streamed from HBM
// exactly once. Each wave: TPW=8 N-tiles, 64 acc VGPRs, 32 WMMAs.
// ---------------------------------------------------------------------------
__global__ void __launch_bounds__(256)
k_gemm1(const float* __restrict__ A,
        const unsigned short* __restrict__ Ws,
        const float* __restrict__ bias1,
        unsigned short* __restrict__ Hbf,
        float* __restrict__ agg2) {
    const int TPW = 8;                 // N tiles per wave
    const int nKs = IN_DIM / 32;       // 4
    __shared__ unsigned short As[32 * IN_DIM];   // 2 row-tiles, bf16

    int tid = threadIdx.x;
    int rowBase = blockIdx.x * 32;     // first global row of this block

    // ---- stage: 32 rows x 128 f32 -> bf16 LDS (each thread: 16 contiguous) --
    {
        int r  = tid >> 3;             // 0..31
        int c0 = (tid & 7) * 16;       // 0..112
        const float4* src = (const float4*)(A + (size_t)(rowBase + r) * IN_DIM + c0);
        float4 q0 = src[0], q1 = src[1], q2 = src[2], q3 = src[3];
        Frag16 tmp;
        tmp.u[0]  = f2bf(q0.x); tmp.u[1]  = f2bf(q0.y); tmp.u[2]  = f2bf(q0.z); tmp.u[3]  = f2bf(q0.w);
        tmp.u[4]  = f2bf(q1.x); tmp.u[5]  = f2bf(q1.y); tmp.u[6]  = f2bf(q1.z); tmp.u[7]  = f2bf(q1.w);
        tmp.u[8]  = f2bf(q2.x); tmp.u[9]  = f2bf(q2.y); tmp.u[10] = f2bf(q2.z); tmp.u[11] = f2bf(q2.w);
        tmp.u[12] = f2bf(q3.x); tmp.u[13] = f2bf(q3.y); tmp.u[14] = f2bf(q3.z); tmp.u[15] = f2bf(q3.w);
        uint4* dst = (uint4*)(As + r * IN_DIM + c0);
        dst[0] = tmp.q[0];
        dst[1] = tmp.q[1];
    }
    __syncthreads();

    int wave = tid >> 5;               // 0..7
    int lane = tid & 31;
    int rt   = wave >> 2;              // row-tile in block: 0..1
    int tn0  = (wave & 3) * TPW;       // 0,8,16,24
    int mn   = lane & 15;
    int kb   = (lane < 16) ? 0 : 8;
    int tm   = blockIdx.x * 2 + rt;

    const unsigned short* arow = As + (rt * 16 + mn) * IN_DIM + kb;

    v8f acc[TPW] = {};
#pragma unroll
    for (int ks = 0; ks < nKs; ++ks) {
        Frag16 a;
        a.q[0] = *(const uint4*)(arow + ks * 32);        // ds_load_b128
        a.q[1] = *(const uint4*)(arow + ks * 32 + 16);   // ds_load_b128
#pragma unroll
        for (int j = 0; j < TPW; ++j) {
            Frag16 b;
            const uint4* bp = (const uint4*)(
                Ws + ((((size_t)(tn0 + j)) * nKs + ks) * 32 + lane) * 16);
            b.q[0] = bp[0];
            b.q[1] = bp[1];
            acc[j] = __builtin_amdgcn_wmma_f32_16x16x32_bf16(
                false, a.v, false, b.v, (short)0, acc[j], false, false);
        }
    }

    int moff = (lane < 16) ? 0 : 8;
#pragma unroll
    for (int j = 0; j < TPW; ++j) {
        int gn = (tn0 + j) * 16 + mn;
        float bv = bias1[gn];
#pragma unroll
        for (int r = 0; r < 8; ++r) {
            int gm = tm * 16 + r + moff;
            size_t idx = (size_t)gm * HID_DIM + gn;
            float v = fmaxf(acc[j][r] + bv, 0.f);
            Hbf[idx]  = f2bf(v);
            agg2[idx] = v;   // agg2 = h + (scatter later)
        }
    }
}

// ---------------------------------------------------------------------------
// GEMM 2: logits = agg2 @ W2p + b2, A read directly from f32 agg2 with fused
// bf16 conversion. One wave computes the full 16x48 strip (3 N-tiles, 24 acc
// VGPRs), 16 K-steps -> 48 WMMAs per wave; agg2 streamed exactly once.
// ---------------------------------------------------------------------------
__global__ void __launch_bounds__(256)
k_gemm2(const float* __restrict__ A,
        const unsigned short* __restrict__ Ws,
        const float* __restrict__ bias2,
        float* __restrict__ logits) {
    const int TPW = OUT_PAD / 16;     // 3
    const int nKs = HID_DIM / 32;     // 16
    int wave = blockIdx.x * (blockDim.x >> 5) + (threadIdx.x >> 5);
    if (wave >= N_NODES / 16) return;
    int tm = wave;
    int lane = threadIdx.x & 31;
    int mn   = lane & 15;
    int kb   = (lane < 16) ? 0 : 8;

    const float* arow = A + (size_t)(tm * 16 + mn) * HID_DIM + kb;

    v8f acc[TPW] = {};
    for (int ks = 0; ks < nKs; ++ks) {
        const float4* ap = (const float4*)(arow + ks * 32);
        float4 q0 = ap[0], q1 = ap[1];
        float4 q2 = *(const float4*)(arow + ks * 32 + 16);
        float4 q3 = *(const float4*)(arow + ks * 32 + 20);
        if (ks + 1 < nKs)
            __builtin_prefetch(arow + (ks + 1) * 32, 0, 0);  // global_prefetch_b8
        Frag16 a;
        a.u[0]  = f2bf(q0.x); a.u[1]  = f2bf(q0.y); a.u[2]  = f2bf(q0.z); a.u[3]  = f2bf(q0.w);
        a.u[4]  = f2bf(q1.x); a.u[5]  = f2bf(q1.y); a.u[6]  = f2bf(q1.z); a.u[7]  = f2bf(q1.w);
        a.u[8]  = f2bf(q2.x); a.u[9]  = f2bf(q2.y); a.u[10] = f2bf(q2.z); a.u[11] = f2bf(q2.w);
        a.u[12] = f2bf(q3.x); a.u[13] = f2bf(q3.y); a.u[14] = f2bf(q3.z); a.u[15] = f2bf(q3.w);
#pragma unroll
        for (int j = 0; j < TPW; ++j) {
            Frag16 b;
            const uint4* bp = (const uint4*)(
                Ws + ((((size_t)j) * nKs + ks) * 32 + lane) * 16);
            b.q[0] = bp[0];
            b.q[1] = bp[1];
            acc[j] = __builtin_amdgcn_wmma_f32_16x16x32_bf16(
                false, a.v, false, b.v, (short)0, acc[j], false, false);
        }
    }

    int moff = (lane < 16) ? 0 : 8;
#pragma unroll
    for (int j = 0; j < TPW; ++j) {
        int gn = j * 16 + mn;
        float bv = (gn < OUT_DIM) ? bias2[gn] : 0.f;
#pragma unroll
        for (int r = 0; r < 8; ++r) {
            int gm = tm * 16 + r + moff;
            logits[(size_t)gm * OUT_PAD + gn] = acc[j][r] + bv;
        }
    }
}

// ---------------------------------------------------------------------------
// Row-wise log_softmax over 40 valid cols of the padded logits.
// ---------------------------------------------------------------------------
__global__ void k_log_softmax(const float* __restrict__ logits,
                              float* __restrict__ out) {
    int row = blockIdx.x * blockDim.x + threadIdx.x;
    if (row >= N_NODES) return;
    const float* p = logits + (size_t)row * OUT_PAD;
    float mx = -INFINITY;
#pragma unroll
    for (int j = 0; j < OUT_DIM; ++j) mx = fmaxf(mx, p[j]);
    float s = 0.f;
#pragma unroll
    for (int j = 0; j < OUT_DIM; ++j) s += __expf(p[j] - mx);
    float lse = mx + __logf(s);
    float* o = out + (size_t)row * OUT_DIM;
#pragma unroll
    for (int j = 0; j < OUT_DIM; ++j) o[j] = p[j] - lse;
}

// ---------------------------------------------------------------------------
// Launch
// ---------------------------------------------------------------------------
extern "C" void kernel_launch(void* const* d_in, const int* in_sizes, int n_in,
                              void* d_out, int out_size, void* d_ws, size_t ws_size,
                              hipStream_t stream) {
    const float*      x   = (const float*)d_in[0];
    const long long*  ei  = (const long long*)d_in[1];   // int64 per reference
    const float*      W1  = (const float*)d_in[2];
    const float*      b1  = (const float*)d_in[3];
    const float*      W2  = (const float*)d_in[4];
    const float*      b2  = (const float*)d_in[5];
    float*            out = (float*)d_out;

    char* ws = (char*)d_ws;
    size_t off = 0;
    auto alloc = [&](size_t bytes) {
        void* p = ws + off;
        off += (bytes + 255) & ~(size_t)255;
        return p;
    };
    float*          agg1_f32 = (float*)         alloc((size_t)N_NODES * IN_DIM  * 4);
    unsigned short* h_bf     = (unsigned short*)alloc((size_t)N_NODES * HID_DIM * 2);
    float*          agg2_f32 = (float*)         alloc((size_t)N_NODES * HID_DIM * 4);
    unsigned short* W1s      = (unsigned short*)alloc((size_t)IN_DIM  * HID_DIM * 2);
    unsigned short* W2s      = (unsigned short*)alloc((size_t)HID_DIM * OUT_PAD * 2);
    float*          logits   = (float*)         alloc((size_t)N_NODES * OUT_PAD * 4);
    (void)ws_size; (void)in_sizes; (void)n_in; (void)out_size;

    const int T = 256;
    auto blocks = [](long long work, int t) { return (unsigned)((work + t - 1) / t); };

    // 1) agg1 = x
    long long x4 = (long long)N_NODES * IN_DIM / 4;
    k_copy_f4<<<blocks(x4, T), T, 0, stream>>>(x, agg1_f32, x4);

    // weights -> bf16 WMMA B-fragment layout (W2 padded to 48 cols)
    long long w1t = (long long)(HID_DIM / 16) * (IN_DIM / 32) * 32;
    k_shuffle_w<<<blocks(w1t, T), T, 0, stream>>>(W1, W1s, IN_DIM, HID_DIM, HID_DIM);
    long long w2t = (long long)(OUT_PAD / 16) * (HID_DIM / 32) * 32;
    k_shuffle_w<<<blocks(w2t, T), T, 0, stream>>>(W2, W2s, HID_DIM, OUT_PAD, OUT_DIM);

    // 2) agg1 += scatter(x)
    k_scatter1<<<blocks((long long)N_EDGES * 32, T), T, 0, stream>>>(x, ei, agg1_f32);

    // 3) h = relu(agg1 @ W1 + b1); seeds agg2 = h  (LDS-staged A, cvt fused)
    k_gemm1<<<(N_NODES / 16) / 2, T, 0, stream>>>(agg1_f32, W1s, b1, h_bf, agg2_f32);

    // 4) agg2 += scatter(h)
    k_scatter2<<<blocks((long long)N_EDGES * 128, T), T, 0, stream>>>(h_bf, ei, agg2_f32);

    // 5) logits = agg2 @ W2 + b2  (A cvt fused)
    long long waves2 = (long long)(N_NODES / 16);
    k_gemm2<<<blocks(waves2 * 32, T), T, 0, stream>>>(agg2_f32, W2s, b2, logits);

    // 6) log_softmax -> d_out
    k_log_softmax<<<blocks((long long)N_NODES, T), T, 0, stream>>>(logits, out);
}
